// SimpleGnnPredictor_11416023073319
// MI455X (gfx1250) — matrix operand order repaired and checked by reading, not compiled, not run on hardware
//
#include <hip/hip_runtime.h>
#include <hip/hip_bf16.h>

#define GNN_N 100000
#define GNN_E 1600000
#define GNN_H 128

typedef __attribute__((ext_vector_type(16))) __bf16 v16bf;
typedef __attribute__((ext_vector_type(2)))  __bf16 v2bf;
typedef __attribute__((ext_vector_type(16))) unsigned short v16u;
typedef __attribute__((ext_vector_type(8)))  float v8f;
typedef __attribute__((ext_vector_type(4)))  int v4i;

// f32 -> bf16 via fptrunc (RNE): lets the backend pick a native cvt if the
// target has one, instead of hand-rolled bit twiddling.
__device__ __forceinline__ unsigned short f2bf(float f) {
  __bf16 b = (__bf16)f;
  return __builtin_bit_cast(unsigned short, b);
}
__device__ __forceinline__ unsigned pack_bf2(float a, float b) {
#if __has_builtin(__builtin_amdgcn_cvt_pk_bf16_f32)
  v2bf r = __builtin_amdgcn_cvt_pk_bf16_f32(a, b);
  return __builtin_bit_cast(unsigned, r);
#else
  v2bf r;
  r[0] = (__bf16)a;
  r[1] = (__bf16)b;
  return __builtin_bit_cast(unsigned, r);
#endif
}

// ---- CDNA5 async global->LDS copy (16B), guarded so the file always builds
__device__ __forceinline__ void async_copy16(const void* gsrc, void* ldst) {
#if __has_builtin(__builtin_amdgcn_global_load_async_to_lds_b128)
  __builtin_amdgcn_global_load_async_to_lds_b128(
      (__attribute__((address_space(1))) v4i*)gsrc,
      (__attribute__((address_space(3))) v4i*)ldst, 0, 0);
#else
  *(uint4*)ldst = *(const uint4*)gsrc;
#endif
}
__device__ __forceinline__ void async_wait() {
#if __has_builtin(__builtin_amdgcn_global_load_async_to_lds_b128)
#if __has_builtin(__builtin_amdgcn_s_wait_asynccnt)
  __builtin_amdgcn_s_wait_asynccnt(0);
#else
  asm volatile("s_wait_asynccnt 0" ::: "memory");
#endif
#endif
}

// ---- WMMA fragment loaders (CDNA5 16x16x32 bf16 layouts, ISA 7.12.2) ----
__device__ __forceinline__ v16bf load_a(const unsigned short* tile, int stride,
                                        int lane, int k0) {
  int m = lane & 15;
  int kh = (lane >> 4) << 3;
  const unsigned short* p = tile + m * stride + k0 + kh;
  v16u t;
#pragma unroll
  for (int v = 0; v < 8; ++v) {
    int kb = (v < 4) ? (v * 2) : (16 + (v - 4) * 2);
    t[2 * v]     = p[kb];
    t[2 * v + 1] = p[kb + 1];
  }
  return __builtin_bit_cast(v16bf, t);
}

__device__ __forceinline__ v16bf load_b(const unsigned short* W, int ldw,
                                        int lane, int k0, int n) {
  int col = n + (lane & 15);
  int kh = (lane >> 4) << 3;
  const unsigned short* p = W + (size_t)(k0 + kh) * ldw + col;
  v16u t;
#pragma unroll
  for (int j = 0; j < 16; ++j) t[j] = p[(size_t)(j < 8 ? j : j + 8) * ldw];
  return __builtin_bit_cast(v16bf, t);
}

__device__ __forceinline__ v8f wmma_bf16(v16bf a, v16bf b, v8f c) {
  return __builtin_amdgcn_wmma_f32_16x16x32_bf16(false, a, false, b, (short)0,
                                                 c, false, false);
}

// Store 8 accumulator values (rows row0..row0+7, one col) as bf16 into a
// row-major [rows][128] LDS tile, pairing rows through a packed convert.
__device__ __forceinline__ void store_c_bf16(unsigned short* tile, int row0,
                                             int col, const v8f& c,
                                             float bias) {
#pragma unroll
  for (int i = 0; i < 4; ++i) {
    float a = fmaxf(c[2 * i] + bias, 0.f);
    float b = fmaxf(c[2 * i + 1] + bias, 0.f);
    unsigned p = pack_bf2(a, b);
    tile[(row0 + 2 * i) * 128 + col] = (unsigned short)p;
    tile[(row0 + 2 * i + 1) * 128 + col] = (unsigned short)(p >> 16);
  }
}

// ---- weight f32 -> bf16 conversion with K padding ----
__global__ void convert_pad_kernel(const float* __restrict__ src,
                                   unsigned short* __restrict__ dst,
                                   int K, int Kpad) {
  int i = blockIdx.x * blockDim.x + threadIdx.x;
  if (i >= Kpad * 128) return;
  int k = i >> 7;
  int j = i & 127;
  dst[i] = (k < K) ? f2bf(src[k * 128 + j]) : (unsigned short)0;
}

// ---- node embed: h = relu(x*W0 + b0), also bf16 mirror ----
__global__ void embed_kernel(const float* __restrict__ x,
                             const float* __restrict__ W0,
                             const float* __restrict__ b0,
                             float* __restrict__ h,
                             unsigned short* __restrict__ hbf) {
  int i = blockIdx.x * blockDim.x + threadIdx.x;
  if (i >= GNN_N * GNN_H) return;
  int n = i >> 7, j = i & 127;
  float v = fmaxf(x[n] * W0[j] + b0[j], 0.f);
  h[i] = v;
  hbf[i] = f2bf(v);
}

__global__ void deg_kernel(const long long* __restrict__ ei,
                           float* __restrict__ deg) {
  int e = blockIdx.x * blockDim.x + threadIdx.x;
  if (e < GNN_E) atomicAdd(&deg[(int)ei[(size_t)GNN_E + e]], 1.0f);
}

// ---- edge MLP + scatter:  agg[dst] += mlp2([h_bf[src] || eattr]) ----
__global__ __launch_bounds__(256) void edge_kernel(
    const unsigned short* __restrict__ hbf, const long long* __restrict__ ei,
    const float* __restrict__ eattr, const unsigned short* __restrict__ W1,
    const float* __restrict__ b1, const unsigned short* __restrict__ W2,
    const float* __restrict__ b2, float* __restrict__ agg, int numTiles) {
  __shared__ unsigned short At[32 * 160];
  __shared__ unsigned short Hid[32 * 128];
  __shared__ int dstS[32];

  int t = threadIdx.x, lane = t & 31, w = t >> 5, n0 = w * 16;
  int e = t >> 3, part = t & 7;  // 32 rows x 8 stager-threads

  v16bf B1[5], B2[4];
#pragma unroll
  for (int kc = 0; kc < 5; ++kc) B1[kc] = load_b(W1, 128, lane, kc * 32, n0);
#pragma unroll
  for (int kc = 0; kc < 4; ++kc) B2[kc] = load_b(W2, 128, lane, kc * 32, n0);
  float bias1 = b1[n0 + (lane & 15)];
  float bias2 = b2[n0 + (lane & 15)];
  int col = n0 + (lane & 15);
  int rbase = (lane >> 4) << 3;

  for (int tile = blockIdx.x; tile < numTiles; tile += gridDim.x) {
    int e0 = tile * 32;
    // --- stage A-tile [32 edges][160 bf16] ---
    long long srcN = ei[e0 + e];
#pragma unroll
    for (int q = 0; q < 2; ++q) {
      int c = 2 * part + q;  // 16B chunk 0..15 of the 128-elem h row
      async_copy16(hbf + (size_t)srcN * GNN_H + c * 8, At + e * 160 + c * 8);
    }
    if (part == 0) {
      dstS[e] = (int)ei[(size_t)GNN_E + e0 + e];
      unsigned* a32 = (unsigned*)(At + e * 160 + 128);
      const float* ea = eattr + (size_t)(e0 + e) * 4;
      a32[0] = pack_bf2(ea[0], ea[1]);
      a32[1] = pack_bf2(ea[2], ea[3]);
#pragma unroll
      for (int i = 2; i < 16; ++i) a32[i] = 0;
    }
    async_wait();
    __syncthreads();

    // --- GEMM1: [32x160] x [160x16], two independent accumulators ---
    v8f c0 = {}, c1 = {};
#pragma unroll
    for (int kc = 0; kc < 5; ++kc) {
      c0 = wmma_bf16(load_a(At, 160, lane, kc * 32), B1[kc], c0);
      c1 = wmma_bf16(load_a(At + 16 * 160, 160, lane, kc * 32), B1[kc], c1);
    }
    store_c_bf16(Hid, rbase, col, c0, bias1);
    store_c_bf16(Hid, 16 + rbase, col, c1, bias1);
    __syncthreads();

    // --- GEMM2: [32x128] x [128x16] ---
    v8f m0 = {}, m1 = {};
#pragma unroll
    for (int kc = 0; kc < 4; ++kc) {
      m0 = wmma_bf16(load_a(Hid, 128, lane, kc * 32), B2[kc], m0);
      m1 = wmma_bf16(load_a(Hid + 16 * 128, 128, lane, kc * 32), B2[kc], m1);
    }

    // --- scatter-add to agg[dst] ---
#pragma unroll
    for (int r = 0; r < 8; ++r) {
      atomicAdd(&agg[(size_t)dstS[rbase + r] * GNN_H + col], m0[r] + bias2);
      atomicAdd(&agg[(size_t)dstS[16 + rbase + r] * GNN_H + col],
                m1[r] + bias2);
    }
    __syncthreads();
  }
}

// ---- node update:  h += mlp2([h || agg/deg]) ; refresh bf16 mirror ----
__global__ __launch_bounds__(256) void node_kernel(
    float* __restrict__ h, unsigned short* __restrict__ hbf,
    const float* __restrict__ agg, const float* __restrict__ deg,
    const unsigned short* __restrict__ W3, const float* __restrict__ b3,
    const unsigned short* __restrict__ W4, const float* __restrict__ b4,
    int numTiles) {
  __shared__ unsigned short At[32 * 256];
  __shared__ unsigned short Hid[32 * 128];
  int t = threadIdx.x, lane = t & 31, w = t >> 5, n0 = w * 16;
  int r = t >> 3, part = t & 7;

  v16bf B3[8], B4[4];
#pragma unroll
  for (int kc = 0; kc < 8; ++kc) B3[kc] = load_b(W3, 128, lane, kc * 32, n0);
#pragma unroll
  for (int kc = 0; kc < 4; ++kc) B4[kc] = load_b(W4, 128, lane, kc * 32, n0);
  float bias3 = b3[n0 + (lane & 15)];
  float bias4 = b4[n0 + (lane & 15)];
  int col = n0 + (lane & 15);
  int rbase = (lane >> 4) << 3;

  for (int tile = blockIdx.x; tile < numTiles; tile += gridDim.x) {
    int nid = tile * 32 + r;
    // h half: async 16B chunks
#pragma unroll
    for (int q = 0; q < 2; ++q) {
      int c = 2 * part + q;
      async_copy16(hbf + (size_t)nid * GNN_H + c * 8, At + r * 256 + c * 8);
    }
    // agg/deg half: needs VALU scaling
    float dg = deg[nid];
    float inv = 1.0f / (dg > 1.0f ? dg : 1.0f);
    const float* arow = agg + (size_t)nid * GNN_H + part * 16;
    unsigned* dsta = (unsigned*)(At + r * 256 + 128) + part * 8;
#pragma unroll
    for (int i = 0; i < 8; ++i)
      dsta[i] = pack_bf2(arow[2 * i] * inv, arow[2 * i + 1] * inv);
    async_wait();
    __syncthreads();

    v8f c0 = {}, c1 = {};
#pragma unroll
    for (int kc = 0; kc < 8; ++kc) {
      c0 = wmma_bf16(load_a(At, 256, lane, kc * 32), B3[kc], c0);
      c1 = wmma_bf16(load_a(At + 16 * 256, 256, lane, kc * 32), B3[kc], c1);
    }
    store_c_bf16(Hid, rbase, col, c0, bias3);
    store_c_bf16(Hid, 16 + rbase, col, c1, bias3);
    __syncthreads();

    v8f o0 = {}, o1 = {};
#pragma unroll
    for (int kc = 0; kc < 4; ++kc) {
      o0 = wmma_bf16(load_a(Hid, 128, lane, kc * 32), B4[kc], o0);
      o1 = wmma_bf16(load_a(Hid + 16 * 128, 128, lane, kc * 32), B4[kc], o1);
    }

#pragma unroll
    for (int rr = 0; rr < 8; ++rr) {
      size_t i0 = (size_t)(tile * 32 + rbase + rr) * GNN_H + col;
      size_t i1 = (size_t)(tile * 32 + 16 + rbase + rr) * GNN_H + col;
      float n0h = h[i0] + o0[rr] + bias4;
      float n1h = h[i1] + o1[rr] + bias4;
      h[i0] = n0h; hbf[i0] = f2bf(n0h);
      h[i1] = n1h; hbf[i1] = f2bf(n1h);
    }
    __syncthreads();
  }
}

// ---- heads: disp = mlp2(out_lin,h), vel = mlp2(vel_lin,h) ----
__global__ __launch_bounds__(256) void head_kernel(
    const unsigned short* __restrict__ hbf, const unsigned short* __restrict__ Wo1,
    const float* __restrict__ bo1, const float* __restrict__ Wo2,
    const float* __restrict__ bo2, const unsigned short* __restrict__ Wv1,
    const float* __restrict__ bv1, const float* __restrict__ Wv2,
    const float* __restrict__ bv2, float* __restrict__ out, int numTiles) {
  __shared__ unsigned short At[32 * 128];
  __shared__ float HidO[32 * 128];
  __shared__ float HidV[32 * 128];
  int t = threadIdx.x, lane = t & 31, w = t >> 5, n0 = w * 16;
  int r = t >> 3, part = t & 7;

  v16bf BO[4], BV[4];
#pragma unroll
  for (int kc = 0; kc < 4; ++kc) BO[kc] = load_b(Wo1, 128, lane, kc * 32, n0);
#pragma unroll
  for (int kc = 0; kc < 4; ++kc) BV[kc] = load_b(Wv1, 128, lane, kc * 32, n0);
  float biasO = bo1[n0 + (lane & 15)];
  float biasV = bv1[n0 + (lane & 15)];
  int col = n0 + (lane & 15);
  int rbase = (lane >> 4) << 3;

  for (int tile = blockIdx.x; tile < numTiles; tile += gridDim.x) {
    int nid = tile * 32 + r;
#pragma unroll
    for (int q = 0; q < 2; ++q) {
      int c = 2 * part + q;
      async_copy16(hbf + (size_t)nid * GNN_H + c * 8, At + r * 128 + c * 8);
    }
    async_wait();
    __syncthreads();

    v8f co0 = {}, co1 = {}, cv0 = {}, cv1 = {};
#pragma unroll
    for (int kc = 0; kc < 4; ++kc) {
      v16bf a0 = load_a(At, 128, lane, kc * 32);
      v16bf a1 = load_a(At + 16 * 128, 128, lane, kc * 32);
      co0 = wmma_bf16(a0, BO[kc], co0);
      co1 = wmma_bf16(a1, BO[kc], co1);
      cv0 = wmma_bf16(a0, BV[kc], cv0);
      cv1 = wmma_bf16(a1, BV[kc], cv1);
    }
#pragma unroll
    for (int rr = 0; rr < 8; ++rr) {
      HidO[(rbase + rr) * 128 + col] = fmaxf(co0[rr] + biasO, 0.f);
      HidO[(16 + rbase + rr) * 128 + col] = fmaxf(co1[rr] + biasO, 0.f);
      HidV[(rbase + rr) * 128 + col] = fmaxf(cv0[rr] + biasV, 0.f);
      HidV[(16 + rbase + rr) * 128 + col] = fmaxf(cv1[rr] + biasV, 0.f);
    }
    __syncthreads();

    if (t < 160) {  // 32 nodes x (3 disp + 2 vel)
      int node = t / 5, oi = t % 5;
      int nn = tile * 32 + node;
      if (oi < 3) {
        float acc = bo2[oi];
        for (int k = 0; k < 128; ++k)
          acc += HidO[node * 128 + k] * Wo2[k * 3 + oi];
        out[(size_t)nn * 3 + oi] = acc;
      } else {
        int oj = oi - 3;
        float acc = bv2[oj];
        for (int k = 0; k < 128; ++k)
          acc += HidV[node * 128 + k] * Wv2[k * 2 + oj];
        out[(size_t)GNN_N * 3 + (size_t)nn * 2 + oj] = acc;
      }
    }
    __syncthreads();
  }
}

extern "C" void kernel_launch(void* const* d_in, const int* in_sizes, int n_in,
                              void* d_out, int out_size, void* d_ws,
                              size_t ws_size, hipStream_t stream) {
  (void)n_in; (void)out_size; (void)ws_size;
  const float* x = (const float*)d_in[0];
  const long long* ei = (const long long*)d_in[1];
  const float* eattr = (const float*)d_in[2];

  // Resolve param-leaf ordering (jax tree-flatten sorts dict keys).
  const float* leaf[22];
  for (int i = 0; i < 22; ++i) leaf[i] = (const float*)d_in[3 + i];
  bool layersFirst = (in_sizes[3] == 132 * 128);
  const float *W1[2], *b1[2], *W2[2], *b2[2], *W3[2], *b3[2], *W4[2], *b4[2];
  const float *W0, *b0, *Wo1, *bo1, *Wo2, *bo2, *Wv1, *bv1, *Wv2, *bv2;
  int idx = 0;
  if (!layersFirst) { W0 = leaf[idx++]; b0 = leaf[idx++]; }
  for (int l = 0; l < 2; ++l) {
    W1[l] = leaf[idx++]; b1[l] = leaf[idx++];
    W2[l] = leaf[idx++]; b2[l] = leaf[idx++];
    W3[l] = leaf[idx++]; b3[l] = leaf[idx++];
    W4[l] = leaf[idx++]; b4[l] = leaf[idx++];
  }
  if (layersFirst) { W0 = leaf[idx++]; b0 = leaf[idx++]; }
  Wo1 = leaf[idx++]; bo1 = leaf[idx++]; Wo2 = leaf[idx++]; bo2 = leaf[idx++];
  Wv1 = leaf[idx++]; bv1 = leaf[idx++]; Wv2 = leaf[idx++]; bv2 = leaf[idx++];

  // Workspace carve-up
  char* ws = (char*)d_ws;
  size_t off = 0;
  auto alloc = [&](size_t bytes) -> void* {
    void* p = ws + off;
    off = (off + bytes + 255) & ~(size_t)255;
    return p;
  };
  float* h = (float*)alloc((size_t)GNN_N * GNN_H * 4);
  unsigned short* hbf = (unsigned short*)alloc((size_t)GNN_N * GNN_H * 2);
  float* agg = (float*)alloc((size_t)GNN_N * GNN_H * 4);
  float* deg = (float*)alloc((size_t)GNN_N * 4);
  unsigned short *W1b[2], *W2b[2], *W3b[2], *W4b[2];
  for (int l = 0; l < 2; ++l) {
    W1b[l] = (unsigned short*)alloc(160 * 128 * 2);
    W2b[l] = (unsigned short*)alloc(128 * 128 * 2);
    W3b[l] = (unsigned short*)alloc(256 * 128 * 2);
    W4b[l] = (unsigned short*)alloc(128 * 128 * 2);
  }
  unsigned short* Wo1b = (unsigned short*)alloc(128 * 128 * 2);
  unsigned short* Wv1b = (unsigned short*)alloc(128 * 128 * 2);

  auto conv = [&](const float* s, unsigned short* d, int K, int Kp) {
    convert_pad_kernel<<<(Kp * 128 + 255) / 256, 256, 0, stream>>>(s, d, K, Kp);
  };
  for (int l = 0; l < 2; ++l) {
    conv(W1[l], W1b[l], 132, 160);
    conv(W2[l], W2b[l], 128, 128);
    conv(W3[l], W3b[l], 256, 256);
    conv(W4[l], W4b[l], 128, 128);
  }
  conv(Wo1, Wo1b, 128, 128);
  conv(Wv1, Wv1b, 128, 128);

  embed_kernel<<<(GNN_N * GNN_H + 255) / 256, 256, 0, stream>>>(x, W0, b0, h,
                                                                hbf);
  (void)hipMemsetAsync(deg, 0, (size_t)GNN_N * 4, stream);
  deg_kernel<<<(GNN_E + 255) / 256, 256, 0, stream>>>(ei, deg);

  const int edgeTiles = GNN_E / 32;  // 50000
  const int nodeTiles = GNN_N / 32;  // 3125
  for (int l = 0; l < 2; ++l) {
    (void)hipMemsetAsync(agg, 0, (size_t)GNN_N * GNN_H * 4, stream);
    edge_kernel<<<4096, 256, 0, stream>>>(hbf, ei, eattr, W1b[l], b1[l],
                                          W2b[l], b2[l], agg, edgeTiles);
    node_kernel<<<2048, 256, 0, stream>>>(h, hbf, agg, deg, W3b[l], b3[l],
                                          W4b[l], b4[l], nodeTiles);
  }
  head_kernel<<<2048, 256, 0, stream>>>(hbf, Wo1b, bo1, Wo2, bo2, Wv1b, bv1,
                                        Wv2, bv2, (float*)d_out, nodeTiles);
}
